// PicoDetHead_82334523064548
// MI455X (gfx1250) — matrix operand order repaired and checked by reading, not compiled, not run on hardware
//
#include <hip/hip_runtime.h>
#include <math.h>

typedef __attribute__((ext_vector_type(16))) _Float16 v16h;
typedef __attribute__((ext_vector_type(8)))  _Float16 v8h;
typedef __attribute__((ext_vector_type(8)))  float    v8f;

#define NTOT    134400L   // 16 * 8400 anchors across all levels
#define NA_IMG  8400
#define NB      16
#define K1      1000
#define MAXDET  100

// Swizzled activation layout: Xs[tile*1536 + n*96 + k]  (tile = 16 anchors, k = channel 0..95)
#define XIDX(col, k) ((((col) >> 4) * 1536) + (((col) & 15) * 96) + (k))

// ---------------------------------------------------------------- helpers

__device__ __forceinline__ float sigmoidf_(float x) { return 1.0f / (1.0f + __expf(-x)); }

// Pre-swizzled A fragment: tile (mt,kt) stored as 32 lanes x 16 contiguous f16 (1KB).
__device__ __forceinline__ v16h load_a_sw(const _Float16* __restrict__ Wsw,
                                          int mt, int kt, int lane) {
  return *(const v16h*)(Wsw + (((mt * 3 + kt) << 9) + (lane << 4)));
}

// B-fragment: lane (n = l&15, h = l>>4), element e holds K = k0 + 16*h + e (contiguous, 32B aligned).
__device__ __forceinline__ v16h load_b_frag_sw(const _Float16* __restrict__ Xs,
                                               long tile, int k0, int lane) {
  int n = lane & 15, h = lane >> 4;
  return *(const v16h*)(Xs + tile * 1536 + n * 96 + k0 + 16 * h);
}

__device__ __forceinline__ void level_info(long n0, long& base, long& HW, int& Wd,
                                           float& stride, int& aBase, int& lvl) {
  if (n0 < 102400) { base = 0;      HW = 6400; Wd = 80; stride = 8.f;  aBase = 0;    lvl = 0; }
  else if (n0 < 128000) { base = 102400; HW = 1600; Wd = 40; stride = 16.f; aBase = 6400; lvl = 1; }
  else { base = 128000; HW = 400;  Wd = 20; stride = 32.f; aBase = 8000; lvl = 2; }
}

struct ClsHeadW { const _Float16* cw[3]; const float* cb[3];
                  const float* ow[3]; const float* ob[3]; };
struct RegHeadW { const _Float16* rw[3]; const float* rb[3]; };

// ---------------------------------------------------------------- weight prep

// Swizzle f32 weights W[M][96] into A-fragment-native f16 tiles.
// Tile (mt,kt): out[(mt*3+kt)*512 + lane*16 + e] = W[mt*16 + (lane&15)][kt*32 + 16*(lane>>4) + e]
__global__ void swizzle_weights(const float* __restrict__ W, _Float16* __restrict__ out,
                                int Mtiles) {
  int id = blockIdx.x * blockDim.x + threadIdx.x;
  if (id >= Mtiles * 3 * 512) return;
  int e    = id & 15;
  int lane = (id >> 4) & 31;
  int tile = id >> 9;
  int mt = tile / 3, kt = tile % 3;
  int m = mt * 16 + (lane & 15);
  int h = lane >> 4;
  int k = kt * 32 + 16 * h + e;
  out[id] = (_Float16)W[m * 96 + k];
}

// Fold BN into per-channel scale/bias (48 channels).
__global__ void fold_bn(const float* __restrict__ g, const float* __restrict__ b,
                        const float* __restrict__ m, const float* __restrict__ v,
                        float* __restrict__ s, float* __restrict__ bb) {
  int i = threadIdx.x;
  if (i < 48) {
    float sc = g[i] * rsqrtf(v[i] + 1e-5f);
    s[i]  = sc;
    bb[i] = b[i] - m[i] * sc;
  }
}

// ---------------------------------------------------------------- convert f32 NCHW -> f16 swizzled

__global__ void convert_feats(const float* __restrict__ f0, const float* __restrict__ f1,
                              const float* __restrict__ f2, _Float16* __restrict__ xs) {
  long id = (long)blockIdx.x * blockDim.x + threadIdx.x;
  if (id >= 96 * NTOT) return;
  int  c   = (int)(id % 96);     // channel fastest -> coalesced f16 writes
  long col = id / 96;
  const float* f; long HW, local;
  if (col < 102400)      { f = f0; HW = 6400; local = col; }
  else if (col < 128000) { f = f1; HW = 1600; local = col - 102400; }
  else                   { f = f2; HW = 400;  local = col - 128000; }
  long b = local / HW, hw = local % HW;
  xs[XIDX(col, c)] = (_Float16)f[(b * 96 + c) * HW + hw];
}

// ---------------------------------------------------------------- ghost primary (1x1 conv 96->48, BN, ReLU6)

__global__ __launch_bounds__(256) void ghost_primary(
    const _Float16* __restrict__ X, _Float16* __restrict__ Y,
    const _Float16* __restrict__ Wsw,
    const float* __restrict__ sc, const float* __restrict__ bi) {
  int lane = threadIdx.x & 31;
  int wid  = threadIdx.x >> 5;
  long tile = (long)blockIdx.x * 8 + wid;       // grid sized exactly: 1050*8 = 8400 tiles

  __builtin_prefetch(X + tile * 1536 + (lane & 15) * 96, 0, 3);

  v8f acc[3] = {};
#pragma unroll
  for (int kt = 0; kt < 3; ++kt) {
    v16h B = load_b_frag_sw(X, tile, kt * 32, lane);
#pragma unroll
    for (int mt = 0; mt < 3; ++mt) {
      v16h A = load_a_sw(Wsw, mt, kt, lane);
      acc[mt] = __builtin_amdgcn_wmma_f32_16x16x32_f16(
          false, A, false, B, (short)0, acc[mt], false, false);
    }
  }

  int h = lane >> 4, n = lane & 15;
  _Float16* lanePtr = Y + tile * 1536 + n * 96;
#pragma unroll
  for (int mt = 0; mt < 3; ++mt) {
    v8h pkt;
#pragma unroll
    for (int i = 0; i < 8; ++i) {
      int m = mt * 16 + 8 * h + i;
      pkt[i] = (_Float16)fminf(fmaxf(acc[mt][i] * sc[m] + bi[m], 0.f), 6.f);
    }
    *(v8h*)(lanePtr + mt * 16 + 8 * h) = pkt;   // 16B aligned contiguous store
  }
}

// ---------------------------------------------------------------- ghost depthwise 3x3 (k 0..47 -> 48..95)

__global__ void ghost_depthwise(_Float16* __restrict__ buf,
    const float* __restrict__ W9,
    const float* __restrict__ scb, const float* __restrict__ bib,
    long base, int H, int Wd, long nPix) {
  long id = (long)blockIdx.x * blockDim.x + threadIdx.x;
  if (id >= 48 * nPix) return;
  long pix = id % nPix;
  int  c   = (int)(id / nPix);
  long HW  = (long)H * Wd;
  long b   = pix / HW;
  long hw  = pix % HW;
  int hh = (int)(hw / Wd), ww = (int)(hw % Wd);
  long colBase = base + b * HW;
  float acc = 0.f;
#pragma unroll
  for (int dy = -1; dy <= 1; ++dy)
#pragma unroll
    for (int dx = -1; dx <= 1; ++dx) {
      int y = hh + dy, x = ww + dx;
      if (y >= 0 && y < H && x >= 0 && x < Wd) {
        long col = colBase + (long)y * Wd + x;
        acc += W9[c * 9 + (dy + 1) * 3 + (dx + 1)] * (float)buf[XIDX(col, c)];
      }
    }
  float v = fminf(fmaxf(acc * scb[c] + bib[c], 0.f), 6.f);
  buf[XIDX(colBase + hw, 48 + c)] = (_Float16)v;
}

// ---------------------------------------------------------------- cls (M=80) + obj head, fused score max/argmax

__global__ __launch_bounds__(256) void head_cls_obj(const _Float16* __restrict__ X,
    ClsHeadW P, float* __restrict__ conf, int* __restrict__ labels) {
  int lane = threadIdx.x & 31;
  int wid  = threadIdx.x >> 5;
  long tile = (long)blockIdx.x * 8 + wid;
  long n0 = tile * 16;
  long base, HW; int Wd, aBase, lvl; float stride;
  level_info(n0, base, HW, Wd, stride, aBase, lvl);
  const _Float16* Wc = P.cw[lvl]; const float* Bc = P.cb[lvl];
  const float*    Wo = P.ow[lvl]; const float* Bo = P.ob[lvl];

  __builtin_prefetch(X + tile * 1536 + (lane & 15) * 96, 0, 3);

  v8f acc[5] = {};
#pragma unroll
  for (int kt = 0; kt < 3; ++kt) {
    v16h B = load_b_frag_sw(X, tile, kt * 32, lane);
#pragma unroll
    for (int mt = 0; mt < 5; ++mt) {
      v16h A = load_a_sw(Wc, mt, kt, lane);
      acc[mt] = __builtin_amdgcn_wmma_f32_16x16x32_f16(
          false, A, false, B, (short)0, acc[mt], false, false);
    }
  }

  // objectness: contiguous 96-f16 dot per anchor column (weights are wave-uniform -> scalar loads)
  int n = lane & 15;
  const _Float16* xp = X + tile * 1536 + n * 96;
  float objv = Bo[0];
  for (int k = 0; k < 96; ++k) objv += Wo[k] * (float)xp[k];

  // max / argmax over 80 class logits (rows split across lane halves)
  int hh = lane >> 4;
  float bv = -3.0e38f; int bi = 0;
#pragma unroll
  for (int t = 0; t < 5; ++t)
#pragma unroll
    for (int i = 0; i < 8; ++i) {
      int m = 16 * t + i + 8 * hh;
      float v = acc[t][i] + Bc[m];
      if (v > bv) { bv = v; bi = m; }
    }
  float ov = __shfl_xor(bv, 16, 32);
  int   oi = __shfl_xor(bi, 16, 32);
  if (ov > bv || (ov == bv && oi < bi)) { bv = ov; bi = oi; }

  if (lane < 16) {
    long ncol  = n0 + n;
    long local = ncol - base;
    long bimg  = local / HW;
    long hw    = local % HW;
    long a     = aBase + hw;
    conf[bimg * NA_IMG + a]   = sigmoidf_(bv) * sigmoidf_(objv);
    labels[bimg * NA_IMG + a] = bi;
  }
}

// ---------------------------------------------------------------- reg head (M=32) + DFL decode -> boxes

__global__ __launch_bounds__(256) void head_reg_decode(const _Float16* __restrict__ X,
    RegHeadW P, float* __restrict__ boxes) {
  __shared__ float lds[8][32][16];
  int lane = threadIdx.x & 31;
  int wid  = threadIdx.x >> 5;
  long tile = (long)blockIdx.x * 8 + wid;
  long n0 = tile * 16;
  long base, HW; int Wd, aBase, lvl; float stride;
  level_info(n0, base, HW, Wd, stride, aBase, lvl);
  const _Float16* Wr = P.rw[lvl]; const float* Br = P.rb[lvl];

  v8f acc[2] = {};
#pragma unroll
  for (int kt = 0; kt < 3; ++kt) {
    v16h B = load_b_frag_sw(X, tile, kt * 32, lane);
#pragma unroll
    for (int mt = 0; mt < 2; ++mt) {
      v16h A = load_a_sw(Wr, mt, kt, lane);
      acc[mt] = __builtin_amdgcn_wmma_f32_16x16x32_f16(
          false, A, false, B, (short)0, acc[mt], false, false);
    }
  }

  int hh = lane >> 4, n = lane & 15;
#pragma unroll
  for (int mt = 0; mt < 2; ++mt)
#pragma unroll
    for (int i = 0; i < 8; ++i) {
      int m = 16 * mt + i + 8 * hh;
      lds[wid][m][n] = acc[mt][i] + Br[m];
    }
  __syncthreads();

  if (lane < 16) {
    long ncol  = n0 + lane;
    long local = ncol - base;
    long bimg  = local / HW;
    long hw    = local % HW;
    int hy = (int)(hw / Wd), wx = (int)(hw % Wd);
    float gx = wx * stride + stride * 0.5f;
    float gy = hy * stride + stride * 0.5f;
    float ltrb[4];
#pragma unroll
    for (int f = 0; f < 4; ++f) {
      float mx = -3.0e38f;
#pragma unroll
      for (int j = 0; j < 8; ++j) mx = fmaxf(mx, lds[wid][8 * f + j][lane]);
      float ssum = 0.f, wsum = 0.f;
#pragma unroll
      for (int j = 0; j < 8; ++j) {
        float e = __expf(lds[wid][8 * f + j][lane] - mx);
        ssum += e; wsum += e * (float)j;
      }
      ltrb[f] = (wsum / ssum) * stride;
    }
    long a = aBase + hw;
    float* ob = boxes + (bimg * NA_IMG + a) * 4;
    ob[0] = gx - ltrb[0];
    ob[1] = gy - ltrb[1];
    ob[2] = gx + ltrb[2];
    ob[3] = gy + ltrb[3];
  }
}

// ---------------------------------------------------------------- per-image NMS (1 workgroup per image)

__global__ __launch_bounds__(256) void nms_kernel(const float* __restrict__ boxes,
    const float* __restrict__ conf, const int* __restrict__ labels,
    float* __restrict__ scws, float* __restrict__ out) {
  __shared__ float ts[K1];
  __shared__ int   tl[K1];
  __shared__ float tb[K1 * 4];
  __shared__ unsigned char keep[K1];
  __shared__ float ks[K1];
  __shared__ float rv[256];
  __shared__ int   ri[256];
  __shared__ int   supFlag;
  __shared__ float maxCoord;

  int b = blockIdx.x, tid = threadIdx.x;
  const float* cf = conf   + (long)b * NA_IMG;
  const float* bx = boxes  + (long)b * NA_IMG * 4;
  const int*   lb = labels + (long)b * NA_IMG;
  float*       sw = scws   + (long)b * NA_IMG;

  for (int a = tid; a < NA_IMG; a += 256) {
    float s = cf[a];
    sw[a] = (s > 0.3f) ? s : -1.0f;
  }
  __syncthreads();

  // top-K1 selection (deterministic ties -> smallest index, matching stable top_k)
  for (int it = 0; it < K1; ++it) {
    float bv = -3.0e38f; int bi = NA_IMG;
    for (int a = tid; a < NA_IMG; a += 256) {
      float s = sw[a];
      if (s > bv || (s == bv && a < bi)) { bv = s; bi = a; }
    }
    rv[tid] = bv; ri[tid] = bi;
    __syncthreads();
    for (int o = 128; o > 0; o >>= 1) {
      if (tid < o) {
        float ov = rv[tid + o]; int oi = ri[tid + o];
        if (ov > rv[tid] || (ov == rv[tid] && oi < ri[tid])) { rv[tid] = ov; ri[tid] = oi; }
      }
      __syncthreads();
    }
    if (tid == 0) {
      int idx = ri[0];
      ts[it] = rv[0];
      tl[it] = lb[idx];
      tb[it * 4 + 0] = bx[idx * 4 + 0]; tb[it * 4 + 1] = bx[idx * 4 + 1];
      tb[it * 4 + 2] = bx[idx * 4 + 2]; tb[it * 4 + 3] = bx[idx * 4 + 3];
      sw[idx] = -3.0e38f;
    }
    __syncthreads();
  }

  // max coordinate among valid boxes (baseline 0, as reference)
  float ml = 0.f;
  for (int i = tid; i < K1; i += 256)
    if (ts[i] > 0.3f) {
      ml = fmaxf(ml, tb[i * 4 + 0]); ml = fmaxf(ml, tb[i * 4 + 1]);
      ml = fmaxf(ml, tb[i * 4 + 2]); ml = fmaxf(ml, tb[i * 4 + 3]);
    }
  rv[tid] = ml; __syncthreads();
  for (int o = 128; o > 0; o >>= 1) { if (tid < o) rv[tid] = fmaxf(rv[tid], rv[tid + o]); __syncthreads(); }
  if (tid == 0) maxCoord = rv[0] + 1.0f;
  for (int i = tid; i < K1; i += 256) keep[i] = 0;
  __syncthreads();

  float mc = maxCoord;
  for (int i = 0; i < K1; ++i) {
    if (tid == 0) supFlag = 0;
    __syncthreads();
    float offi = (float)tl[i] * mc;
    float ix1 = tb[i * 4 + 0] + offi, iy1 = tb[i * 4 + 1] + offi;
    float ix2 = tb[i * 4 + 2] + offi, iy2 = tb[i * 4 + 3] + offi;
    float areaI = fmaxf(ix2 - ix1, 0.f) * fmaxf(iy2 - iy1, 0.f);
    int local = 0;
    for (int j = tid; j < i; j += 256) {
      if (!keep[j]) continue;
      float offj = (float)tl[j] * mc;
      float jx1 = tb[j * 4 + 0] + offj, jy1 = tb[j * 4 + 1] + offj;
      float jx2 = tb[j * 4 + 2] + offj, jy2 = tb[j * 4 + 3] + offj;
      float areaJ = fmaxf(jx2 - jx1, 0.f) * fmaxf(jy2 - jy1, 0.f);
      float xx1 = fmaxf(ix1, jx1), yy1 = fmaxf(iy1, jy1);
      float xx2 = fminf(ix2, jx2), yy2 = fminf(iy2, jy2);
      float inter = fmaxf(xx2 - xx1, 0.f) * fmaxf(yy2 - yy1, 0.f);
      float iou = inter / fmaxf(areaI + areaJ - inter, 1e-9f);
      if (iou > 0.5f) local = 1;
    }
    if (local) atomicOr(&supFlag, 1);
    __syncthreads();
    if (tid == 0) keep[i] = (ts[i] > 0.3f) && (supFlag == 0);
    __syncthreads();
  }

  for (int i = tid; i < K1; i += 256) ks[i] = keep[i] ? ts[i] : -3.0e38f;
  __syncthreads();

  float* outB = out;
  float* outS = out + (long)NB * MAXDET * 4;
  float* outL = outS + (long)NB * MAXDET;
  for (int slot = 0; slot < MAXDET; ++slot) {
    float bv = -3.0e38f; int bi = K1;
    for (int i = tid; i < K1; i += 256) {
      float s = ks[i];
      if (s > bv || (s == bv && i < bi)) { bv = s; bi = i; }
    }
    rv[tid] = bv; ri[tid] = bi;
    __syncthreads();
    for (int o = 128; o > 0; o >>= 1) {
      if (tid < o) {
        float ov = rv[tid + o]; int oi = ri[tid + o];
        if (ov > rv[tid] || (ov == rv[tid] && oi < ri[tid])) { rv[tid] = ov; ri[tid] = oi; }
      }
      __syncthreads();
    }
    if (tid == 0) {
      int idx = ri[0];
      bool ok = rv[0] > -1.0e38f;
      long ob = (long)b * MAXDET + slot;
      if (ok) {
        outB[ob * 4 + 0] = tb[idx * 4 + 0]; outB[ob * 4 + 1] = tb[idx * 4 + 1];
        outB[ob * 4 + 2] = tb[idx * 4 + 2]; outB[ob * 4 + 3] = tb[idx * 4 + 3];
        outS[ob] = ts[idx]; outL[ob] = (float)tl[idx];
        ks[idx] = -3.0e38f;
      } else {
        outB[ob * 4 + 0] = 0.f; outB[ob * 4 + 1] = 0.f;
        outB[ob * 4 + 2] = 0.f; outB[ob * 4 + 3] = 0.f;
        outS[ob] = 0.f; outL[ob] = -1.0f;
      }
    }
    __syncthreads();
  }
}

// ---------------------------------------------------------------- host

extern "C" void kernel_launch(void* const* d_in, const int* in_sizes, int n_in,
                              void* d_out, int out_size, void* d_ws, size_t ws_size,
                              hipStream_t stream) {
  (void)in_sizes; (void)n_in; (void)out_size; (void)ws_size;
  const float* f0 = (const float*)d_in[0];
  const float* f1 = (const float*)d_in[1];
  const float* f2 = (const float*)d_in[2];
  // ghost param layout per module: pw,pg,pb,pm,pv,cw,cg,cb,cm,cv
  auto G = [&](int baseIdx, int j) { return (const float*)d_in[baseIdx + j]; };
  // module bases: cls_tower ghost0 @3, ghost1 @13; reg_tower ghost0 @23, ghost1 @33
  const int gbase[4] = {3, 13, 23, 33};

  char* ws = (char*)d_ws;
  size_t off = 0;
  auto alloc = [&](size_t bytes) -> void* {
    off = (off + 255) & ~(size_t)255;
    void* p = ws + off;
    off += bytes;
    return p;
  };
  _Float16* xin  = (_Float16*)alloc(96 * NTOT * sizeof(_Float16));
  _Float16* bufA = (_Float16*)alloc(96 * NTOT * sizeof(_Float16));
  _Float16* bufB = (_Float16*)alloc(96 * NTOT * sizeof(_Float16));
  float* conf    = (float*)alloc(NTOT * sizeof(float));
  int*   labels  = (int*)alloc(NTOT * sizeof(int));
  float* boxes   = (float*)alloc(NTOT * 4 * sizeof(float));
  float* scws    = (float*)alloc(NTOT * sizeof(float));
  // swizzled f16 weights: ghost 4 x 9 tiles, cls 3 x 15 tiles, reg 3 x 6 tiles (512 f16 per tile)
  _Float16* gW[4]; for (int g = 0; g < 4; ++g) gW[g] = (_Float16*)alloc(9  * 512 * sizeof(_Float16));
  _Float16* cW[3]; for (int l = 0; l < 3; ++l) cW[l] = (_Float16*)alloc(15 * 512 * sizeof(_Float16));
  _Float16* rW[3]; for (int l = 0; l < 3; ++l) rW[l] = (_Float16*)alloc(6  * 512 * sizeof(_Float16));
  // folded BN: per ghost module: primary (s,b) + depthwise (s,b), 48 each
  float* fbn[4][4];
  for (int g = 0; g < 4; ++g)
    for (int j = 0; j < 4; ++j) fbn[g][j] = (float*)alloc(48 * sizeof(float));

  dim3 blk(256);

  // ---- prep: swizzle weights + fold BN (tiny, every call, deterministic)
  for (int g = 0; g < 4; ++g) {
    swizzle_weights<<<(9 * 512 + 255) / 256, blk, 0, stream>>>(G(gbase[g], 0), gW[g], 3);
    fold_bn<<<1, 64, 0, stream>>>(G(gbase[g],1), G(gbase[g],2), G(gbase[g],3), G(gbase[g],4),
                                  fbn[g][0], fbn[g][1]);
    fold_bn<<<1, 64, 0, stream>>>(G(gbase[g],6), G(gbase[g],7), G(gbase[g],8), G(gbase[g],9),
                                  fbn[g][2], fbn[g][3]);
  }
  ClsHeadW CH; RegHeadW RH;
  for (int l = 0; l < 3; ++l) {
    swizzle_weights<<<(15 * 512 + 255) / 256, blk, 0, stream>>>((const float*)d_in[43 + 2 * l], cW[l], 5);
    swizzle_weights<<<(6  * 512 + 255) / 256, blk, 0, stream>>>((const float*)d_in[55 + 2 * l], rW[l], 2);
    CH.cw[l] = cW[l];
    CH.cb[l] = (const float*)d_in[44 + 2 * l];
    CH.ow[l] = (const float*)d_in[49 + 2 * l];
    CH.ob[l] = (const float*)d_in[50 + 2 * l];
    RH.rw[l] = rW[l];
    RH.rb[l] = (const float*)d_in[56 + 2 * l];
  }

  long nconv = 96 * NTOT;
  convert_feats<<<(unsigned)((nconv + 255) / 256), blk, 0, stream>>>(f0, f1, f2, xin);

  struct Lvl { long base; int H, W; long nPix; };
  const Lvl L[3] = { {0, 80, 80, 102400}, {102400, 40, 40, 25600}, {128000, 20, 20, 6400} };

  auto tower_stage = [&](const _Float16* in, _Float16* outp, int g) {
    ghost_primary<<<1050, blk, 0, stream>>>(in, outp, gW[g], fbn[g][0], fbn[g][1]);
    for (int l = 0; l < 3; ++l) {
      long tot = 48 * L[l].nPix;
      ghost_depthwise<<<(unsigned)((tot + 255) / 256), blk, 0, stream>>>(outp,
          G(gbase[g], 5), fbn[g][2], fbn[g][3],
          L[l].base, L[l].H, L[l].W, L[l].nPix);
    }
  };

  // cls tower -> heads
  tower_stage(xin,  bufA, 0);
  tower_stage(bufA, bufB, 1);
  head_cls_obj<<<1050, blk, 0, stream>>>(bufB, CH, conf, labels);

  // reg tower -> reg head + DFL decode
  tower_stage(xin,  bufA, 2);
  tower_stage(bufA, bufB, 3);
  head_reg_decode<<<1050, blk, 0, stream>>>(bufB, RH, boxes);

  nms_kernel<<<NB, blk, 0, stream>>>(boxes, conf, labels, scws, (float*)d_out);
}
